// PointNetSetAbstraction_43087111913598
// MI455X (gfx1250) — matrix operand order repaired and checked by reading, not compiled, not run on hardware
//
#include <hip/hip_runtime.h>

typedef __bf16 bf16_t;
typedef __attribute__((ext_vector_type(8)))  bf16_t v8bf;
typedef __attribute__((ext_vector_type(16))) bf16_t v16bf;
typedef __attribute__((ext_vector_type(8)))  float  v8f;

#define N_PTS    8192
#define N_BATCH  8
#define S_PT     2048
#define K_SMP    32
#define C_PTS    64
#define M_TOT    (N_BATCH * S_PT * K_SMP)   /* 524288 rows through the MLP */
#define INV_M    (1.0f / (float)M_TOT)
#define EPSV     1e-5f
#define R2       0.04f                      /* radius^2 = 0.2^2 */
#define OUT_XYZ  (N_BATCH * S_PT * 3)       /* new_xyz floats in d_out */

/* ---- workspace layout (bytes) ---- */
#define OFF_IDX   ((size_t)0)                                       /* int[8*2048*32] */
#define OFF_NXYZ  ((size_t)(N_BATCH * S_PT * K_SMP) * 4)            /* float[8*2048*3] */
#define OFF_SUMS  (OFF_NXYZ + (size_t)(N_BATCH * S_PT * 3) * 4)     /* float[6*128]  */
#define OFF_Z0    ((OFF_SUMS + (size_t)(6 * 128) * 4 + 255) & ~(size_t)255)
#define OFF_Z1    (OFF_Z0 + (size_t)M_TOT * 64 * 4)
#define OFF_Z2    (OFF_Z1 + (size_t)M_TOT * 64 * 4)

/* =========================================================================
 * WMMA fragment loaders (CDNA5 16-bit layouts, ISA 7.12.2, wave32)
 *  A 16x32: lane<16 -> row=lane, K in {0..7}U{16..23}; lane>=16 -> K+8
 *           => two contiguous 8-element (16B) runs per lane.
 *  B 32x16: lane holds col n=lane&15; K same as A (two 8-element runs).
 * ========================================================================= */
__device__ __forceinline__ v16bf load_fragA(const bf16_t* rowBase, int half) {
  v8bf lo = *(const v8bf*)(rowBase + half * 8);
  v8bf hi = *(const v8bf*)(rowBase + 16 + half * 8);
  return __builtin_shufflevector(lo, hi, 0,1,2,3,4,5,6,7,8,9,10,11,12,13,14,15);
}
__device__ __forceinline__ v16bf load_fragB(const bf16_t* rowBase, int half) {
  v8bf lo = *(const v8bf*)(rowBase + half * 8);
  v8bf hi = *(const v8bf*)(rowBase + 16 + half * 8);
  return __builtin_shufflevector(lo, hi, 0,1,2,3,4,5,6,7,8,9,10,11,12,13,14,15);
}

#define WMMA_BF16(A, Bv, C) \
  __builtin_amdgcn_wmma_f32_16x16x32_bf16(false, (A), false, (Bv), (short)0, (C), false, false)

/* Block = 256 threads = 8 waves, each wave owns a 32x32 output tile.
 * NCH=64 : waves as 4(m) x 2(n)  => block tile 128 x 64
 * NCH=128: waves as 2(m) x 4(n)  => block tile  64 x 128
 * At: MT x KPAD bf16 (row-major), Wt: NCH x KPAD bf16 (W[n][k] == B[k][n]). */
template <int KPAD, int NCH>
__device__ __forceinline__ void wmma_block(const bf16_t* At, const bf16_t* Wt,
                                           float* zOut, int m0,
                                           float* lSum, float* lSq) {
  constexpr int WN = NCH / 32;
  const int tid  = threadIdx.x;
  const int wave = tid >> 5, lane = tid & 31;
  const int wn = wave % WN, wm = wave / WN;
  const int half = lane >> 4, nl = lane & 15;

  v8f acc[2][2] = {};
#pragma unroll
  for (int kt = 0; kt < KPAD / 32; ++kt) {
    const bf16_t* a0p = At + (wm * 32 + nl) * KPAD + kt * 32;
    const bf16_t* b0p = Wt + (wn * 32 + nl) * KPAD + kt * 32;
    v16bf a0 = load_fragA(a0p, half);
    v16bf a1 = load_fragA(a0p + 16 * KPAD, half);
    v16bf b0 = load_fragB(b0p, half);
    v16bf b1 = load_fragB(b0p + 16 * KPAD, half);
    acc[0][0] = WMMA_BF16(a0, b0, acc[0][0]);
    acc[0][1] = WMMA_BF16(a0, b1, acc[0][1]);
    acc[1][0] = WMMA_BF16(a1, b0, acc[1][0]);
    acc[1][1] = WMMA_BF16(a1, b1, acc[1][1]);
  }
#pragma unroll
  for (int tm = 0; tm < 2; ++tm)
#pragma unroll
    for (int tn = 0; tn < 2; ++tn) {
      const int mBase = m0 + wm * 32 + tm * 16 + half * 8;  /* D: m = v + 8*half */
      const int nIdx  = wn * 32 + tn * 16 + nl;             /* D: n = lane&15    */
      float* zp = zOut + (size_t)mBase * NCH + nIdx;
      v8f cc = acc[tm][tn];
      float s = 0.f, q = 0.f;
#pragma unroll
      for (int v = 0; v < 8; ++v) {
        float val = cc[v];
        zp[(size_t)v * NCH] = val;
        s += val; q += val * val;
      }
      s += __shfl_down(s, 16, 32);   /* fold lanes 16..31 (same n) */
      q += __shfl_down(q, 16, 32);
      if (half == 0) { atomicAdd(&lSum[nIdx], s); atomicAdd(&lSq[nIdx], q); }
    }
}

/* =========================== FPS ========================================= */
__global__ __launch_bounds__(1024) void fps_kernel(const float* __restrict__ xyz,
                                                   float* __restrict__ out,
                                                   float* __restrict__ wsNewXyz) {
  extern __shared__ float smem[];
  float* xs = smem;            /* 8192 */
  float* ys = xs + N_PTS;
  float* zs = ys + N_PTS;
  float* dist = zs + N_PTS;    /* 8192 */
  float* rv = dist + N_PTS;    /* 1024 */
  int*   ri = (int*)(rv + 1024);
  __shared__ int selS;

  const int b = blockIdx.x, tid = threadIdx.x;
  const float* base = xyz + (size_t)b * N_PTS * 3;
  for (int i = tid; i < N_PTS; i += 1024) {
    xs[i] = base[3 * i]; ys[i] = base[3 * i + 1]; zs[i] = base[3 * i + 2];
    dist[i] = 1e10f;
  }
  __syncthreads();

  int sel = 0;
  for (int it = 0; it < S_PT; ++it) {
    if (tid == 0) {
      size_t o = ((size_t)b * S_PT + it) * 3;
      float px = xs[sel], py = ys[sel], pz = zs[sel];
      out[o] = px; out[o + 1] = py; out[o + 2] = pz;
      wsNewXyz[o] = px; wsNewXyz[o + 1] = py; wsNewXyz[o + 2] = pz;
    }
    if (it == S_PT - 1) break;
    const float cx = xs[sel], cy = ys[sel], cz = zs[sel];
    float best = -1.0f; int bi = 0;
    for (int i = tid; i < N_PTS; i += 1024) {
      float dx = xs[i] - cx, dy = ys[i] - cy, dz = zs[i] - cz;
      float dd = fminf(dist[i], dx * dx + dy * dy + dz * dz);
      dist[i] = dd;
      if (dd > best) { best = dd; bi = i; }
    }
    rv[tid] = best; ri[tid] = bi;
    __syncthreads();
    for (int off = 512; off > 0; off >>= 1) {
      if (tid < off) {
        float ov = rv[tid + off]; int oi = ri[tid + off];
        if (ov > rv[tid] || (ov == rv[tid] && oi < ri[tid])) { rv[tid] = ov; ri[tid] = oi; }
      }
      __syncthreads();
    }
    if (tid == 0) selS = ri[0];
    __syncthreads();
    sel = selS;
  }
}

/* ======================== Ball query ===================================== */
__global__ __launch_bounds__(512) void ballquery_kernel(const float* __restrict__ xyz,
                                                        const float* __restrict__ newxyz,
                                                        int* __restrict__ idxOut) {
  extern __shared__ float smem[];
  float* xs = smem; float* ys = xs + N_PTS; float* zs = ys + N_PTS;
  const int b = blockIdx.x >> 7;                /* 128 blocks / batch */
  const int sBase = (blockIdx.x & 127) * 16;    /* 16 waves = 16 centroids */
  const float* base = xyz + (size_t)b * N_PTS * 3;
  for (int i = threadIdx.x; i < N_PTS; i += 512) {
    xs[i] = base[3 * i]; ys[i] = base[3 * i + 1]; zs[i] = base[3 * i + 2];
  }
  __syncthreads();

  const int wave = threadIdx.x >> 5, lane = threadIdx.x & 31;
  const int s = sBase + wave;
  const size_t so = (size_t)b * S_PT + s;
  const float cx = newxyz[so * 3], cy = newxyz[so * 3 + 1], cz = newxyz[so * 3 + 2];
  int* o = idxOut + so * K_SMP;

  int found = 0, firstIdx = -1;
  for (int ch = 0; ch < N_PTS / 32 && found < K_SMP; ++ch) {
    const int j = ch * 32 + lane;
    const float dx = xs[j] - cx, dy = ys[j] - cy, dz = zs[j] - cz;
    const bool in = (dx * dx + dy * dy + dz * dz) <= R2;
    const unsigned mask = __builtin_amdgcn_ballot_w32(in);
    if (firstIdx < 0 && mask) firstIdx = ch * 32 + __ffs((int)mask) - 1;
    const int pos = found + __popc(mask & ((1u << lane) - 1u));
    if (in && pos < K_SMP) o[pos] = j;
    found += __popc(mask);
  }
  if (firstIdx < 0) firstIdx = 0;
  for (int p = (found < K_SMP ? found : K_SMP) + lane; p < K_SMP; p += 32) o[p] = firstIdx;
}

/* ============ Layer 0: fused gather/concat + GEMM (K=67 pad 96) ========== */
__global__ __launch_bounds__(256) void l0_kernel(const int* __restrict__ idx,
                                                 const float* __restrict__ wsNewXyz,
                                                 const float* __restrict__ xyz,
                                                 const float* __restrict__ pts,
                                                 const float* __restrict__ w0,
                                                 float* __restrict__ z0,
                                                 float* __restrict__ gSum, float* __restrict__ gSq) {
  __shared__ __align__(16) bf16_t At[128 * 96];
  __shared__ __align__(16) bf16_t Wt[64 * 96];
  __shared__ float lSum[64], lSq[64];
  const int tid = threadIdx.x;
  const int m0 = blockIdx.x * 128;

  for (int i = tid; i < 64 * 96; i += 256) {
    int n = i / 96, c = i - n * 96;
    Wt[i] = (c < 67) ? (bf16_t)w0[n * 67 + c] : (bf16_t)0.0f;
  }
  if (tid < 64) { lSum[tid] = 0.f; lSq[tid] = 0.f; }

  { /* gather + concat one half-row per thread */
    const int mloc = tid >> 1, chalf = tid & 1;
    const int m = m0 + mloc;
    const int b = m >> 16;               /* 32*2048 rows per batch */
    const int rem = m & 65535;
    const int s = rem >> 5, kk = rem & 31;
    const int p = idx[(((size_t)b * S_PT + s) << 5) + kk];
    const float* nx = wsNewXyz + ((size_t)b * S_PT + s) * 3;
    const float* px = xyz + ((size_t)b * N_PTS + p) * 3;
    const float* pf = pts + ((size_t)b * N_PTS + p) * (size_t)C_PTS;
    bf16_t* row = At + mloc * 96;
    if (chalf == 0) {
      row[0] = (bf16_t)(px[0] - nx[0]);
      row[1] = (bf16_t)(px[1] - nx[1]);
      row[2] = (bf16_t)(px[2] - nx[2]);
      for (int c = 0; c < 45; ++c) row[3 + c] = (bf16_t)pf[c];
    } else {
      for (int c = 45; c < 64; ++c) row[3 + c] = (bf16_t)pf[c];
      for (int c = 67; c < 96; ++c) row[c] = (bf16_t)0.0f;
    }
  }
  __syncthreads();
  wmma_block<96, 64>(At, Wt, z0, m0, lSum, lSq);
  __syncthreads();
  if (tid < 64) { atomicAdd(&gSum[tid], lSum[tid]); atomicAdd(&gSq[tid], lSq[tid]); }
}

/* ====== Layers 1/2: fused BN(prev)+ReLU -> bf16 A staging -> GEMM ======== */
template <int NCH>
__global__ __launch_bounds__(256) void lN_kernel(const float* __restrict__ zPrev,
                                                 const float* __restrict__ w,
                                                 const float* __restrict__ gammaP,
                                                 const float* __restrict__ betaP,
                                                 const float* __restrict__ sumP,
                                                 const float* __restrict__ sqP,
                                                 float* __restrict__ zOut,
                                                 float* __restrict__ gSum, float* __restrict__ gSq) {
  constexpr int MT = (NCH == 64) ? 128 : 64;
  __shared__ __align__(16) bf16_t At[MT * 64];
  __shared__ __align__(16) bf16_t Wt[NCH * 64];
  __shared__ float lSum[NCH], lSq[NCH], aS[64], aB[64];
  const int tid = threadIdx.x;
  const int m0 = blockIdx.x * MT;

  if (tid < 64) {
    float mean = sumP[tid] * INV_M;
    float var  = sqP[tid] * INV_M - mean * mean;
    float sc   = rsqrtf(var + EPSV) * gammaP[tid];
    aS[tid] = sc; aB[tid] = betaP[tid] - mean * sc;
  }
  if (tid < NCH) { lSum[tid] = 0.f; lSq[tid] = 0.f; }
  for (int i = tid; i < NCH * 64; i += 256) Wt[i] = (bf16_t)w[i];
  __syncthreads();
  for (int i = tid; i < MT * 64; i += 256) {
    const int c = i & 63;
    const float zv = zPrev[(size_t)m0 * 64 + i];   /* coalesced stream */
    At[i] = (bf16_t)fmaxf(zv * aS[c] + aB[c], 0.0f);
  }
  __syncthreads();
  wmma_block<64, NCH>(At, Wt, zOut, m0, lSum, lSq);
  __syncthreads();
  if (tid < NCH) { atomicAdd(&gSum[tid], lSum[tid]); atomicAdd(&gSq[tid], lSq[tid]); }
}

/* ============== Final: BN + ReLU + max over nsample ====================== */
__global__ __launch_bounds__(256) void pool_kernel(const float* __restrict__ z2,
                                                   const float* __restrict__ sum2,
                                                   const float* __restrict__ sq2,
                                                   const float* __restrict__ g2,
                                                   const float* __restrict__ b2,
                                                   float* __restrict__ out) {
  const int t = blockIdx.x * 256 + threadIdx.x;   /* 8*2048*128 threads */
  const int c = t & 127;
  const size_t bs = (size_t)(t >> 7);
  const float mean = sum2[c] * INV_M;
  const float var  = sq2[c] * INV_M - mean * mean;
  const float sc = rsqrtf(var + EPSV) * g2[c];
  const float sb = b2[c] - mean * sc;
  const float* base = z2 + bs * K_SMP * 128 + c;
  float m = 0.0f;                                  /* relu => max >= 0 */
#pragma unroll 4
  for (int kk = 0; kk < K_SMP; ++kk)
    m = fmaxf(m, fmaxf(base[(size_t)kk * 128] * sc + sb, 0.0f));
  out[OUT_XYZ + bs * 128 + c] = m;
}

__global__ void zero_sums_kernel(float* sums) {
  if (threadIdx.x < 6 * 128 / 2) {
    sums[2 * threadIdx.x] = 0.f; sums[2 * threadIdx.x + 1] = 0.f;
  }
}

/* ========================================================================= */
extern "C" void kernel_launch(void* const* d_in, const int* in_sizes, int n_in,
                              void* d_out, int out_size, void* d_ws, size_t ws_size,
                              hipStream_t stream) {
  (void)in_sizes; (void)n_in; (void)out_size; (void)ws_size;
  const float* xyz = (const float*)d_in[0];
  const float* pts = (const float*)d_in[1];
  const float* w0  = (const float*)d_in[2];
  const float* g0  = (const float*)d_in[3];
  const float* bt0 = (const float*)d_in[4];
  const float* w1  = (const float*)d_in[5];
  const float* g1  = (const float*)d_in[6];
  const float* bt1 = (const float*)d_in[7];
  const float* w2  = (const float*)d_in[8];
  const float* g2  = (const float*)d_in[9];
  const float* bt2 = (const float*)d_in[10];
  float* out = (float*)d_out;

  char* ws = (char*)d_ws;
  int*   wsIdx  = (int*)(ws + OFF_IDX);
  float* wsNXyz = (float*)(ws + OFF_NXYZ);
  float* sums   = (float*)(ws + OFF_SUMS);
  float *sum0 = sums,        *sq0 = sums + 128;
  float *sum1 = sums + 256,  *sq1 = sums + 384;
  float *sum2 = sums + 512,  *sq2 = sums + 640;
  float* z0 = (float*)(ws + OFF_Z0);
  float* z1 = (float*)(ws + OFF_Z1);
  float* z2 = (float*)(ws + OFF_Z2);

  const size_t fpsLds = (size_t)(4 * N_PTS) * 4 + 1024 * 4 + 1024 * 4; /* 139264 B */
  const size_t bqLds  = (size_t)(3 * N_PTS) * 4;                       /*  98304 B */

  zero_sums_kernel<<<1, 512, 0, stream>>>(sums);
  fps_kernel<<<N_BATCH, 1024, fpsLds, stream>>>(xyz, out, wsNXyz);
  ballquery_kernel<<<N_BATCH * (S_PT / 16), 512, bqLds, stream>>>(xyz, wsNXyz, wsIdx);
  l0_kernel<<<M_TOT / 128, 256, 0, stream>>>(wsIdx, wsNXyz, xyz, pts, w0, z0, sum0, sq0);
  lN_kernel<64> <<<M_TOT / 128, 256, 0, stream>>>(z0, w1, g0, bt0, sum0, sq0, z1, sum1, sq1);
  lN_kernel<128><<<M_TOT / 64,  256, 0, stream>>>(z1, w2, g1, bt1, sum1, sq1, z2, sum2, sq2);
  pool_kernel<<<(N_BATCH * S_PT * 128) / 256, 256, 0, stream>>>(z2, sum2, sq2, g2, bt2, out);
}
